// GeneratorEncDecTeacherForcingV3_54623394070922
// MI455X (gfx1250) — matrix operand order repaired and checked by reading, not compiled, not run on hardware
//
#include <hip/hip_runtime.h>
#include <math.h>

// Problem dims (hard-coded per reference)
#define V_SZ 50257
#define E_SZ 512
#define H_SZ 1024
#define LM_SZ 128
#define LC_SZ 256

// Persistent recurrent kernel geometry
#define NWG 16
#define TPB 256
#define GT (NWG * TPB)

// GEMM M-register-blocking: each wave computes MBLK 16x16 tiles stacked in M,
// sharing one B fragment per K-step (amortizes f32->bf16 conversion of the
// weight side, which is the huge matrix in the dominant out_W GEMM).
#define MBLK 4

typedef __bf16 bf16_t;
typedef bf16_t v16bf __attribute__((ext_vector_type(16)));
typedef float v8f __attribute__((ext_vector_type(8)));

union BF16Frag {
  v16bf v;
  bf16_t e[16];
};

// Let the backend lower fptrunc f32->bf16 to the native convert ops
// (gfx1250 is bf16-native; avoids 3-VALU/element bit-math emulation).
__device__ __forceinline__ void pack4(bf16_t* d, float4 f) {
  d[0] = (bf16_t)f.x;
  d[1] = (bf16_t)f.y;
  d[2] = (bf16_t)f.z;
  d[3] = (bf16_t)f.w;
}

// D[m][n] = sum_k A[m][k] * W[n][k] + bias[n]
// A rows optionally gathered via Aidx (embedding lookup fused into fragment
// load). M = gridDim.y*16*MBLK (exact), N may be ragged (guarded), K % 32 == 0.
__global__ __launch_bounds__(32) void wmma_gemm_rowmajor(
    const float* __restrict__ Atab, const int* __restrict__ Aidx, int lda,
    const float* __restrict__ W, int ldw,
    const float* __restrict__ bias,
    float* __restrict__ D, int ldd, int N, int K) {
  const int lane = threadIdx.x;
  const int l15 = lane & 15;
  const bool hi = lane >= 16;
  const int m0 = blockIdx.y * (16 * MBLK);
  const int n0 = blockIdx.x << 4;

  const float* arow[MBLK];
#pragma unroll
  for (int mi = 0; mi < MBLK; ++mi) {
    const int mrow = m0 + mi * 16 + l15;
    arow[mi] = Aidx ? (Atab + (size_t)Aidx[mrow] * lda)
                    : (Atab + (size_t)mrow * lda);
  }
  int nrow = n0 + l15;
  if (nrow > N - 1) nrow = N - 1;  // clamp ragged edge; store is guarded
  const float* brow = W + (size_t)nrow * ldw;

  // ISA 7.12.2 16-bit A 16x32 / B 32x16 per-lane K offsets
  const int aoff = hi ? 8 : 0;
  const int boff = hi ? 8 : 0;

  v8f acc[MBLK];
#pragma unroll
  for (int mi = 0; mi < MBLK; ++mi)
    acc[mi] = (v8f){0.f, 0.f, 0.f, 0.f, 0.f, 0.f, 0.f, 0.f};

  for (int k0 = 0; k0 < K; k0 += 32) {
    BF16Frag b;
    pack4(b.e + 0,  *(const float4*)(brow + k0 + boff));
    pack4(b.e + 4,  *(const float4*)(brow + k0 + boff + 4));
    pack4(b.e + 8,  *(const float4*)(brow + k0 + 16 + boff));
    pack4(b.e + 12, *(const float4*)(brow + k0 + 16 + boff + 4));
#pragma unroll
    for (int mi = 0; mi < MBLK; ++mi) {
      BF16Frag a;
      pack4(a.e + 0,  *(const float4*)(arow[mi] + k0 + aoff));
      pack4(a.e + 4,  *(const float4*)(arow[mi] + k0 + aoff + 4));
      pack4(a.e + 8,  *(const float4*)(arow[mi] + k0 + 16 + aoff));
      pack4(a.e + 12, *(const float4*)(arow[mi] + k0 + 16 + aoff + 4));
      acc[mi] = __builtin_amdgcn_wmma_f32_16x16x32_bf16(
          false, a.v, false, b.v, (short)0, acc[mi], false, false);
    }
  }

  const int nn = n0 + l15;
  if (nn < N) {
    const float bs = bias ? bias[nn] : 0.0f;
#pragma unroll
    for (int mi = 0; mi < MBLK; ++mi) {
      const int mb = m0 + mi * 16 + (hi ? 8 : 0);
#pragma unroll
      for (int r = 0; r < 8; ++r)
        D[(size_t)(mb + r) * ldd + nn] = acc[mi][r] + bs;
    }
  }
}

// ---------------- persistent recurrent kernel --------------------------------

__device__ __forceinline__ float sigmoidf_(float x) {
  return 1.0f / (1.0f + __expf(-x));
}

// s: LDS-staged activation vector, g: global weight row (16B aligned)
__device__ __forceinline__ float dotv(const float* __restrict__ s,
                                      const float* __restrict__ g, int n) {
  float acc = 0.f;
  for (int i = 0; i < n; i += 4) {
    float4 w = *(const float4*)(g + i);
    acc = fmaf(s[i + 0], w.x, acc);
    acc = fmaf(s[i + 1], w.y, acc);
    acc = fmaf(s[i + 2], w.z, acc);
    acc = fmaf(s[i + 3], w.w, acc);
  }
  return acc;
}

__device__ __forceinline__ void gsync(unsigned* bar, unsigned* ep) {
  __syncthreads();
  if (threadIdx.x == 0) {
    __threadfence();  // release stores from this WG
    const unsigned target = (++(*ep)) * NWG;
    atomicAdd(bar, 1u);
    while (__hip_atomic_load(bar, __ATOMIC_ACQUIRE, __HIP_MEMORY_SCOPE_AGENT) <
           target)
      __builtin_amdgcn_s_sleep(1);
    __threadfence();  // acquire: invalidate WGP cache before re-reads
  }
  __syncthreads();
}

__global__ __launch_bounds__(TPB) void recurrent_kernel(
    float* __restrict__ h, float* __restrict__ gh, float* __restrict__ gi,
    float* __restrict__ alog, float* __restrict__ ctx, float* __restrict__ x,
    const float* __restrict__ Gi_enc, float* __restrict__ enc_outs,
    const float* __restrict__ Pre_attn, const float* __restrict__ Pre_comb,
    float* __restrict__ H2, unsigned* __restrict__ bar,
    const float* __restrict__ enc_Whh, const float* __restrict__ enc_bhh,
    const float* __restrict__ dec_Whh, const float* __restrict__ dec_bhh,
    const float* __restrict__ dec_Wih, const float* __restrict__ dec_bih,
    const float* __restrict__ attn_W, const float* __restrict__ comb_W) {
  __shared__ float sv[H_SZ];
  __shared__ float sa[LM_SZ];
  const int tid = threadIdx.x;
  const int gtid = blockIdx.x * TPB + tid;
  unsigned ep = 0;

  // ----- encoder: 128 sequential GRU steps; gi precomputed by WMMA GEMM -----
  for (int t = 0; t < LM_SZ; ++t) {
    for (int i = tid; i < H_SZ; i += TPB) sv[i] = h[i];
    __syncthreads();
    for (int j = gtid; j < 3 * H_SZ; j += GT)
      gh[j] = dotv(sv, enc_Whh + (size_t)j * H_SZ, H_SZ) + enc_bhh[j];
    gsync(bar, &ep);
    const float* git = Gi_enc + (size_t)t * 3 * H_SZ;
    for (int i = gtid; i < H_SZ; i += GT) {
      float r = sigmoidf_(git[i] + gh[i]);
      float z = sigmoidf_(git[H_SZ + i] + gh[H_SZ + i]);
      float n = tanhf(git[2 * H_SZ + i] + r * gh[2 * H_SZ + i]);
      float hn = (1.0f - z) * n + z * h[i];
      h[i] = hn;
      enc_outs[(size_t)t * H_SZ + i] = hn;
    }
    gsync(bar, &ep);
  }

  // ----- decoder: 256 sequential steps with attention ------------------------
  for (int t = 0; t < LC_SZ; ++t) {
    for (int i = tid; i < H_SZ; i += TPB) sv[i] = h[i];
    __syncthreads();
    // attention logits (h-half; emb-half + attn_b precomputed) and gh=h@Whh^T
    for (int j = gtid; j < LM_SZ + 3 * H_SZ; j += GT) {
      if (j < LM_SZ)
        alog[j] = Pre_attn[(size_t)t * LM_SZ + j] +
                  dotv(sv, attn_W + (size_t)j * (2 * H_SZ) + H_SZ, H_SZ);
      else {
        int r = j - LM_SZ;
        gh[r] = dotv(sv, dec_Whh + (size_t)r * H_SZ, H_SZ) + dec_bhh[r];
      }
    }
    gsync(bar, &ep);
    // ctx[i] = softmax(alog) @ enc_outs[:, i]  (softmax fused per thread)
    for (int i = tid; i < LM_SZ; i += TPB) sa[i] = alog[i];
    __syncthreads();
    for (int i = gtid; i < H_SZ; i += GT) {
      float mx = -INFINITY;
      for (int j = 0; j < LM_SZ; ++j) mx = fmaxf(mx, sa[j]);
      float den = 0.f, num = 0.f;
      for (int j = 0; j < LM_SZ; ++j) {
        float e = __expf(sa[j] - mx);
        den += e;
        num = fmaf(e, enc_outs[(size_t)j * H_SZ + i], num);
      }
      ctx[i] = num / den;
    }
    gsync(bar, &ep);
    // x = Pre_comb(emb-half + comb_b) + ctx @ comb_W[:, H:]^T
    for (int i = tid; i < H_SZ; i += TPB) sv[i] = ctx[i];
    __syncthreads();
    for (int i = gtid; i < H_SZ; i += GT)
      x[i] = Pre_comb[(size_t)t * H_SZ + i] +
             dotv(sv, comb_W + (size_t)i * (2 * H_SZ) + H_SZ, H_SZ);
    gsync(bar, &ep);
    // gi = x @ Wih^T + bih
    for (int i = tid; i < H_SZ; i += TPB) sv[i] = x[i];
    __syncthreads();
    for (int j = gtid; j < 3 * H_SZ; j += GT)
      gi[j] = dotv(sv, dec_Wih + (size_t)j * H_SZ, H_SZ) + dec_bih[j];
    gsync(bar, &ep);
    // GRU gate combine; stash h2 into H2 for the batched output GEMM
    for (int i = gtid; i < H_SZ; i += GT) {
      float r = sigmoidf_(gi[i] + gh[i]);
      float z = sigmoidf_(gi[H_SZ + i] + gh[H_SZ + i]);
      float n = tanhf(gi[2 * H_SZ + i] + r * gh[2 * H_SZ + i]);
      float hn = (1.0f - z) * n + z * h[i];
      h[i] = hn;
      H2[(size_t)t * H_SZ + i] = hn;
    }
    gsync(bar, &ep);
  }
}

// ---------------- small kernels ----------------------------------------------

__global__ void init_kernel(const int* __restrict__ tgt, int* __restrict__ dec_in,
                            float* __restrict__ h, unsigned* __restrict__ bar) {
  const int i = threadIdx.x;  // 256 threads
  if (i == 0) {
    dec_in[0] = 0;            // SOS
    dec_in[1] = tgt[LC_SZ - 1];
    *bar = 0;
  }
  if (i >= 2) dec_in[i] = tgt[i - 2];
  for (int j = i; j < H_SZ; j += 256) h[j] = 0.f;
}

__global__ __launch_bounds__(256) void log_softmax_rows(float* __restrict__ d,
                                                        int n) {
  __shared__ float red[256];
  float* p = d + (size_t)blockIdx.x * n;
  const int tid = threadIdx.x;
  float mx = -INFINITY;
  for (int i = tid; i < n; i += 256) mx = fmaxf(mx, p[i]);
  red[tid] = mx;
  __syncthreads();
  for (int s = 128; s > 0; s >>= 1) {
    if (tid < s) red[tid] = fmaxf(red[tid], red[tid + s]);
    __syncthreads();
  }
  mx = red[0];
  __syncthreads();
  float sum = 0.f;
  for (int i = tid; i < n; i += 256) sum += __expf(p[i] - mx);
  red[tid] = sum;
  __syncthreads();
  for (int s = 128; s > 0; s >>= 1) {
    if (tid < s) red[tid] += red[tid + s];
    __syncthreads();
  }
  const float lse = mx + __logf(red[0]);
  __syncthreads();
  for (int i = tid; i < n; i += 256) p[i] -= lse;
}

// ---------------- host launcher ----------------------------------------------

extern "C" void kernel_launch(void* const* d_in, const int* in_sizes, int n_in,
                              void* d_out, int out_size, void* d_ws,
                              size_t ws_size, hipStream_t stream) {
  (void)in_sizes; (void)n_in; (void)out_size; (void)ws_size;
  const int* motion    = (const int*)d_in[0];
  const int* tgt       = (const int*)d_in[1];
  const float* enc_emb = (const float*)d_in[2];
  const float* enc_Wih = (const float*)d_in[3];
  const float* enc_Whh = (const float*)d_in[4];
  const float* enc_bih = (const float*)d_in[5];
  const float* enc_bhh = (const float*)d_in[6];
  const float* dec_emb = (const float*)d_in[7];
  const float* dec_Wih = (const float*)d_in[8];
  const float* dec_Whh = (const float*)d_in[9];
  const float* dec_bih = (const float*)d_in[10];
  const float* dec_bhh = (const float*)d_in[11];
  const float* attn_W  = (const float*)d_in[12];
  const float* attn_b  = (const float*)d_in[13];
  const float* comb_W  = (const float*)d_in[14];
  const float* comb_b  = (const float*)d_in[15];
  const float* out_W   = (const float*)d_in[16];
  const float* out_b   = (const float*)d_in[17];
  float* out = (float*)d_out;

  // workspace carve-up (~4.5 MB total), 256B aligned slots
  char* ws = (char*)d_ws;
  size_t off = 0;
  auto take = [&](size_t bytes) -> char* {
    char* p = ws + off;
    off = (off + bytes + 255) & ~(size_t)255;
    return p;
  };
  unsigned* bar   = (unsigned*)take(256);
  int* dec_in     = (int*)take(LC_SZ * sizeof(int));
  float* h        = (float*)take(H_SZ * 4);
  float* gh       = (float*)take(3 * H_SZ * 4);
  float* gi       = (float*)take(3 * H_SZ * 4);
  float* alog     = (float*)take(LM_SZ * 4);
  float* ctx      = (float*)take(H_SZ * 4);
  float* xbuf     = (float*)take(H_SZ * 4);
  float* Gi_enc   = (float*)take((size_t)LM_SZ * 3 * H_SZ * 4);
  float* enc_outs = (float*)take((size_t)LM_SZ * H_SZ * 4);
  float* Pre_attn = (float*)take((size_t)LC_SZ * LM_SZ * 4);
  float* Pre_comb = (float*)take((size_t)LC_SZ * H_SZ * 4);
  float* H2       = (float*)take((size_t)LC_SZ * H_SZ * 4);

  init_kernel<<<1, 256, 0, stream>>>(tgt, dec_in, h, bar);

  const dim3 b32(32);
  // Gi_enc[128 x 3072] = gather(enc_emb, motion) @ enc_Wih^T + enc_bih
  wmma_gemm_rowmajor<<<dim3(3 * H_SZ / 16, LM_SZ / (16 * MBLK)), b32, 0,
                       stream>>>(enc_emb, motion, E_SZ, enc_Wih, E_SZ, enc_bih,
                                 Gi_enc, 3 * H_SZ, 3 * H_SZ, E_SZ);
  // Pre_attn[256 x 128] = gather(dec_emb, dec_in) @ attn_W[:, :H]^T + attn_b
  wmma_gemm_rowmajor<<<dim3(LM_SZ / 16, LC_SZ / (16 * MBLK)), b32, 0, stream>>>(
      dec_emb, dec_in, H_SZ, attn_W, 2 * H_SZ, attn_b, Pre_attn, LM_SZ, LM_SZ,
      H_SZ);
  // Pre_comb[256 x 1024] = gather(dec_emb, dec_in) @ comb_W[:, :H]^T + comb_b
  wmma_gemm_rowmajor<<<dim3(H_SZ / 16, LC_SZ / (16 * MBLK)), b32, 0, stream>>>(
      dec_emb, dec_in, H_SZ, comb_W, 2 * H_SZ, comb_b, Pre_comb, H_SZ, H_SZ,
      H_SZ);

  // sequential encoder+decoder recurrence (persistent, grid-synced)
  recurrent_kernel<<<NWG, TPB, 0, stream>>>(
      h, gh, gi, alog, ctx, xbuf, Gi_enc, enc_outs, Pre_attn, Pre_comb, H2,
      bar, enc_Whh, enc_bhh, dec_Whh, dec_bhh, dec_Wih, dec_bih, attn_W,
      comb_W);

  // logits[256 x 50257] = H2 @ out_W^T + out_b  (single pass over out_W)
  wmma_gemm_rowmajor<<<dim3((V_SZ + 15) / 16, LC_SZ / (16 * MBLK)), b32, 0,
                       stream>>>(H2, nullptr, H_SZ, out_W, H_SZ, out_b, out,
                                 V_SZ, V_SZ, H_SZ);
  // in-place log_softmax per row
  log_softmax_rows<<<LC_SZ, 256, 0, stream>>>(out, V_SZ);
}
